// MSLSTMCell_21088289423707
// MI455X (gfx1250) — matrix operand order repaired and checked by reading, not compiled, not run on hardware
//
#include <hip/hip_runtime.h>
#include <cstdint>
#include <cstddef>

typedef __attribute__((ext_vector_type(16))) __bf16 v16bf;
typedef __attribute__((ext_vector_type(8)))  __bf16 v8bf;
typedef __attribute__((ext_vector_type(8)))  float  v8f;

namespace {
constexpr int T_STEPS = 128;
constexpr int BATCH   = 256;
constexpr int NIN     = 512;
constexpr int NMS     = 256;
constexpr int HID     = 1024;
constexpr int GATES   = 4 * HID;           // 4096
constexpr int MROWS   = T_STEPS * BATCH;   // 32768

// Phase-1 GEMM tiling: 128x128 block tile, 8 waves of 64x32, K-step 32 (bf16)
constexpr int BM = 128, BN = 128, BK = 32;
constexpr int LDT = BK + 16;               // 48 bf16 = 96B stride: 16B aligned, staggers banks
// Fused-step tiling: 64 rows x (32 h-cols x 4 gates), 8 waves = (2 m-halves) x (4 gates)
constexpr int BMS = 64;
constexpr int HB  = 32;
}

// ---- CDNA5 async global->LDS staging (ASYNCcnt path, bypasses VGPRs) ----
__device__ __forceinline__ void async_load_b128(__bf16* lds_dst, const __bf16* gsrc) {
  // LDS aperture: low 32 bits of a generic LDS address are the LDS byte offset.
  uint32_t lds_off = (uint32_t)(uintptr_t)lds_dst;
  asm volatile("global_load_async_to_lds_b128 %0, %1, off"
               :: "v"(lds_off), "v"(gsrc) : "memory");
}
__device__ __forceinline__ void wait_async0() {
  asm volatile("s_wait_asynccnt 0x0" ::: "memory");
}

// ---------------- tile staging: global (row-major, row length K) -> LDS ----------------
template <int ROWS>
__device__ __forceinline__ void stage_tile(__bf16* dst, const __bf16* __restrict__ src,
                                           int row0, int K, int k0) {
  constexpr int ITERS = (ROWS * 4) / 256;  // 8-bf16 chunks per row = 4
#pragma unroll
  for (int it = 0; it < ITERS; ++it) {
    int id = threadIdx.x + it * 256;
    int r  = id >> 2;
    int c  = (id & 3) << 3;
    async_load_b128(dst + r * LDT + c, src + (size_t)(row0 + r) * K + k0 + c);
  }
}

// B tile for the fused step: 4 gate-strided 32-row blocks of Whh rows g*HID + h0 + n
__device__ __forceinline__ void stage_tile_gates(__bf16* dst, const __bf16* __restrict__ src,
                                                 int h0, int K, int k0) {
#pragma unroll
  for (int it = 0; it < 2; ++it) {
    int id = threadIdx.x + it * 256;
    int r  = id >> 2;                       // 0..127 -> gate = r>>5, n = r&31
    int c  = (id & 3) << 3;
    int grow = (r >> 5) * HID + h0 + (r & 31);
    async_load_b128(dst + r * LDT + c, src + (size_t)grow * K + k0 + c);
  }
}

// A fragment 16x32 bf16: lane L holds row M=L%16; lanes 0-15: K 0..7 & 16..23, lanes 16-31: K 8..15 & 24..31
__device__ __forceinline__ v16bf frag_a(const __bf16* As, int m0, int lane) {
  const int m  = m0 + (lane & 15);
  const int k0 = (lane >> 4) << 3;
  const __bf16* p = As + m * LDT + k0;
  v8bf lo = *reinterpret_cast<const v8bf*>(p);
  v8bf hi = *reinterpret_cast<const v8bf*>(p + 16);
  v16bf r;
#pragma unroll
  for (int i = 0; i < 8; ++i) { r[i] = lo[i]; r[i + 8] = hi[i]; }
  return r;
}

// B fragment 32x16 bf16 (LDS staged transposed [n][k]): lane L holds col N=L%16;
// lanes 0-15: K 0..7 & 16..23, lanes 16-31: K 8..15 & 24..31
__device__ __forceinline__ v16bf frag_b(const __bf16* Bs, int n0, int lane) {
  const int n  = n0 + (lane & 15);
  const int k0 = (lane >> 4) << 3;
  const __bf16* p = Bs + n * LDT + k0;
  v8bf lo = *reinterpret_cast<const v8bf*>(p);
  v8bf hi = *reinterpret_cast<const v8bf*>(p + 16);
  v16bf r;
#pragma unroll
  for (int i = 0; i < 8; ++i) { r[i] = lo[i]; r[i + 8] = hi[i]; }
  return r;
}

// ---------------- Phase 1: pre = X*Wih^T + MS*Wnh^T + (b_ih + b_nh), stored bf16 (NT) ----------------
// Double-buffered async pipeline over a fused 24-tile K loop (16 tiles X/Wih + 8 tiles MS/Wnh).
__global__ __launch_bounds__(256) void gemm_pre_kernel(
    const __bf16* __restrict__ Xbf, const __bf16* __restrict__ Wih,
    const __bf16* __restrict__ MSbf, const __bf16* __restrict__ Wnh,
    const float* __restrict__ bias, __bf16* __restrict__ pre) {
  __shared__ __bf16 As[2][BM * LDT];
  __shared__ __bf16 Bs[2][BN * LDT];
  const int bn = blockIdx.x, bm = blockIdx.y;
  const int lane = threadIdx.x & 31, wid = threadIdx.x >> 5;
  const int wave_m = wid >> 2, wave_n = wid & 3;
  v8f acc[4][2];
#pragma unroll
  for (int mi = 0; mi < 4; ++mi)
#pragma unroll
    for (int ni = 0; ni < 2; ++ni)
#pragma unroll
      for (int r = 0; r < 8; ++r) acc[mi][ni][r] = 0.0f;

  constexpr int NK1 = NIN / BK;   // 16
  constexpr int NK  = NK1 + NMS / BK;  // 24

  auto issue = [&](int t, int buf) {
    if (t < NK1) {
      stage_tile<BM>(As[buf], Xbf,  bm * BM, NIN, t * BK);
      stage_tile<BN>(Bs[buf], Wih,  bn * BN, NIN, t * BK);
    } else {
      stage_tile<BM>(As[buf], MSbf, bm * BM, NMS, (t - NK1) * BK);
      stage_tile<BN>(Bs[buf], Wnh,  bn * BN, NMS, (t - NK1) * BK);
    }
  };

  issue(0, 0);
  for (int t = 0; t < NK; ++t) {
    wait_async0();          // tile t landed in LDS
    __syncthreads();        // visible to all; everyone done reading buf (t+1)&1
    if (t + 1 < NK) issue(t + 1, (t + 1) & 1);   // overlap next tile with compute
    const __bf16* Ab = As[t & 1];
    const __bf16* Bb = Bs[t & 1];
    v16bf a[4], b[2];
#pragma unroll
    for (int mi = 0; mi < 4; ++mi) a[mi] = frag_a(Ab, wave_m * 64 + mi * 16, lane);
#pragma unroll
    for (int ni = 0; ni < 2; ++ni) b[ni] = frag_b(Bb, wave_n * 32 + ni * 16, lane);
#pragma unroll
    for (int mi = 0; mi < 4; ++mi)
#pragma unroll
      for (int ni = 0; ni < 2; ++ni)
        acc[mi][ni] = __builtin_amdgcn_wmma_f32_16x16x32_bf16(
            false, a[mi], false, b[ni], (short)0, acc[mi][ni], false, false);
  }

  // C/D layout: VGPR r, lanes 0-15 -> row r, lanes 16-31 -> row r+8; col = lane%16
  const int col0 = bn * BN + wave_n * 32 + (lane & 15);
  const int row0 = bm * BM + wave_m * 64 + ((lane >> 4) << 3);
#pragma unroll
  for (int mi = 0; mi < 4; ++mi)
#pragma unroll
    for (int ni = 0; ni < 2; ++ni) {
      const int c = col0 + ni * 16;
      const float bsum = bias[c];
#pragma unroll
      for (int r = 0; r < 8; ++r) {
        const int m = row0 + mi * 16 + r;
        __builtin_nontemporal_store((__bf16)(acc[mi][ni][r] + bsum),
                                    pre + (size_t)m * GATES + c);
      }
    }
}

// ---------------- Phase 2: fused h*Whh^T + pre[t] + b_hh -> LSTM cell ----------------
__device__ __forceinline__ float sigf(float x) { return 1.0f / (1.0f + __expf(-x)); }

__global__ __launch_bounds__(256) void fused_step_kernel(
    const __bf16* __restrict__ Hin,   // [256][1024] bf16, h(t)
    const __bf16* __restrict__ Whh,   // [4096][1024] bf16 (L2-resident)
    const __bf16* __restrict__ pre_t, // [256][4096] bf16
    const float* __restrict__ bhh,    // [4096]
    float* __restrict__ cws,          // [256][1024] fp32, in-place
    __bf16* __restrict__ Hout,        // h(t+1) bf16 (ping-pong)
    float* __restrict__ out_t, float* __restrict__ hx_out, float* __restrict__ cx_out) {
  __shared__ __bf16 As[2][BMS * LDT];     // 2 x (64 x 48)
  __shared__ __bf16 Bs[2][128 * LDT];     // 2 x (4 gates x 32 rows)
  __shared__ float  Gs[4 * BMS * HB];     // 32 KB: i/f/g/o for 64x32
  const int h0 = blockIdx.x * HB;
  const int bm = blockIdx.y;
  const int lane = threadIdx.x & 31, wid = threadIdx.x >> 5;
  const int gate = wid & 3, mhalf = wid >> 2;

  v8f acc[2][2];
#pragma unroll
  for (int mi = 0; mi < 2; ++mi)
#pragma unroll
    for (int ni = 0; ni < 2; ++ni)
#pragma unroll
      for (int r = 0; r < 8; ++r) acc[mi][ni][r] = 0.0f;

  constexpr int NK = HID / BK;  // 32
  auto issue = [&](int t, int buf) {
    stage_tile<BMS>(As[buf], Hin, bm * BMS, HID, t * BK);
    stage_tile_gates(Bs[buf], Whh, h0, HID, t * BK);
  };

  issue(0, 0);
  for (int t = 0; t < NK; ++t) {
    wait_async0();
    __syncthreads();
    if (t + 1 < NK) issue(t + 1, (t + 1) & 1);
    const __bf16* Ab = As[t & 1];
    const __bf16* Bb = Bs[t & 1];
    v16bf a[2], b[2];
#pragma unroll
    for (int mi = 0; mi < 2; ++mi) a[mi] = frag_a(Ab, mhalf * 32 + mi * 16, lane);
#pragma unroll
    for (int ni = 0; ni < 2; ++ni) b[ni] = frag_b(Bb, gate * 32 + ni * 16, lane);
#pragma unroll
    for (int mi = 0; mi < 2; ++mi)
#pragma unroll
      for (int ni = 0; ni < 2; ++ni)
        acc[mi][ni] = __builtin_amdgcn_wmma_f32_16x16x32_bf16(
            false, a[mi], false, b[ni], (short)0, acc[mi][ni], false, false);
  }

  // Park gate pre-activations in LDS (add pre[t] + b_hh here)
  const int colL = lane & 15;
  const int rsub = (lane >> 4) << 3;
#pragma unroll
  for (int mi = 0; mi < 2; ++mi)
#pragma unroll
    for (int ni = 0; ni < 2; ++ni) {
      const int col  = ni * 16 + colL;            // 0..31
      const int gcol = gate * HID + h0 + col;     // column in [0,4096)
      const float badd = bhh[gcol];
#pragma unroll
      for (int r = 0; r < 8; ++r) {
        const int row = mhalf * 32 + mi * 16 + rsub + r;   // 0..63
        const int m   = bm * BMS + row;
        const float p = (float)__builtin_nontemporal_load(pre_t + (size_t)m * GATES + gcol);
        Gs[(gate * BMS + row) * HB + col] = acc[mi][ni][r] + p + badd;
      }
    }
  __syncthreads();

  // LSTM cell pointwise on the 64x32 tile (8 elements/thread)
#pragma unroll
  for (int e = 0; e < 8; ++e) {
    const int idx = e * 256 + threadIdx.x;       // 0..2047
    const int row = idx >> 5, col = idx & (HB - 1);
    const float iv = sigf (Gs[(0 * BMS + row) * HB + col]);
    const float fv = sigf (Gs[(1 * BMS + row) * HB + col]);
    const float gv = tanhf(Gs[(2 * BMS + row) * HB + col]);
    const float ov = sigf (Gs[(3 * BMS + row) * HB + col]);
    const size_t gi = (size_t)(bm * BMS + row) * HID + h0 + col;
    const float c  = cws[gi] * fv + iv * gv;
    const float hv = ov * tanhf(c);
    cws[gi]  = c;
    Hout[gi] = (__bf16)hv;                        // next step's WMMA A-operand
    __builtin_nontemporal_store(hv, out_t + gi);  // streaming output
    hx_out[gi] = hv;                              // final step's value survives
    cx_out[gi] = c;
  }
}

// ---------------- prep ----------------
__global__ __launch_bounds__(256) void f32_to_bf16_kernel(const float* __restrict__ s,
                                                          __bf16* __restrict__ d, int n) {
  int i = (blockIdx.x * 256 + threadIdx.x) * 4;
  if (i + 3 < n) {
    float4 v = *reinterpret_cast<const float4*>(s + i);
    d[i]     = (__bf16)v.x;
    d[i + 1] = (__bf16)v.y;
    d[i + 2] = (__bf16)v.z;
    d[i + 3] = (__bf16)v.w;
  } else {
    for (; i < n; ++i) d[i] = (__bf16)s[i];
  }
}

__global__ __launch_bounds__(256) void bias_add_kernel(const float* __restrict__ a,
                                                       const float* __restrict__ b,
                                                       float* __restrict__ o, int n) {
  const int i = blockIdx.x * 256 + threadIdx.x;
  if (i < n) o[i] = a[i] + b[i];
}

extern "C" void kernel_launch(void* const* d_in, const int* in_sizes, int n_in,
                              void* d_out, int out_size, void* d_ws, size_t ws_size,
                              hipStream_t stream) {
  (void)in_sizes; (void)n_in; (void)out_size; (void)ws_size;
  const float* X   = (const float*)d_in[0];
  const float* MS  = (const float*)d_in[1];
  const float* hx  = (const float*)d_in[2];
  const float* cx  = (const float*)d_in[3];
  const float* Wih = (const float*)d_in[4];
  const float* bih = (const float*)d_in[5];
  const float* Whh = (const float*)d_in[6];
  const float* bhh = (const float*)d_in[7];
  const float* Wnh = (const float*)d_in[8];
  const float* bnh = (const float*)d_in[9];
  float* out = (float*)d_out;

  char* base = (char*)d_ws;
  size_t off = 0;
  auto carve = [&](size_t bytes) -> char* {
    char* p = base + off;
    off += (bytes + 255) & ~(size_t)255;
    return p;
  };
  __bf16* Xbf   = (__bf16*)carve((size_t)MROWS * NIN * 2);   // 32 MB
  __bf16* MSbf  = (__bf16*)carve((size_t)MROWS * NMS * 2);   // 16 MB
  __bf16* Wihb  = (__bf16*)carve((size_t)GATES * NIN * 2);   //  4 MB
  __bf16* Wnhb  = (__bf16*)carve((size_t)GATES * NMS * 2);   //  2 MB
  __bf16* Whhb  = (__bf16*)carve((size_t)GATES * HID * 2);   //  8 MB (L2-resident)
  float*  biasp = (float*) carve((size_t)GATES * 4);
  __bf16* pre   = (__bf16*)carve((size_t)MROWS * GATES * 2); // 256 MB (NT streamed)
  __bf16* hb0   = (__bf16*)carve((size_t)BATCH * HID * 2);   // h ping
  __bf16* hb1   = (__bf16*)carve((size_t)BATCH * HID * 2);   // h pong
  float*  cws   = (float*) carve((size_t)BATCH * HID * 4);

  auto conv = [&](const float* s, __bf16* d, int n) {
    int q = (n + 3) / 4;
    f32_to_bf16_kernel<<<(q + 255) / 256, 256, 0, stream>>>(s, d, n);
  };
  conv(X,   Xbf,  MROWS * NIN);
  conv(MS,  MSbf, MROWS * NMS);
  conv(Wih, Wihb, GATES * NIN);
  conv(Wnh, Wnhb, GATES * NMS);
  conv(Whh, Whhb, GATES * HID);
  conv(hx,  hb0,  BATCH * HID);
  bias_add_kernel<<<GATES / 256, 256, 0, stream>>>(bih, bnh, biasp, GATES);
  hipMemcpyAsync(cws, cx, (size_t)BATCH * HID * 4, hipMemcpyDeviceToDevice, stream);

  // Phase 1: all-timestep input projection (32x256 = 8192 workgroups)
  gemm_pre_kernel<<<dim3(GATES / BN, MROWS / BM), 256, 0, stream>>>(
      Xbf, Wihb, MSbf, Wnhb, biasp, pre);

  // Phase 2: sequential recurrence, fused GEMM+cell, ping-pong h buffers (128 WGs/step)
  float* hx_out = out + (size_t)T_STEPS * BATCH * HID;
  float* cx_out = hx_out + (size_t)BATCH * HID;
  __bf16* hbuf[2] = {hb0, hb1};
  for (int t = 0; t < T_STEPS; ++t) {
    fused_step_kernel<<<dim3(HID / HB, BATCH / BMS), 256, 0, stream>>>(
        hbuf[t & 1], Whhb, pre + (size_t)t * BATCH * GATES, bhh, cws,
        hbuf[(t + 1) & 1], out + (size_t)t * BATCH * HID, hx_out, cx_out);
  }
}
